// Basicblock2_35029753266761
// MI455X (gfx1250) — matrix operand order repaired and checked by reading, not compiled, not run on hardware
//
#include <hip/hip_runtime.h>
#include <hip/hip_bf16.h>

// ---------------------------------------------------------------------------
// Types for CDNA5 WMMA (wave32, v_wmma_f32_16x16x32_bf16)
// ---------------------------------------------------------------------------
typedef __bf16 v16bf __attribute__((ext_vector_type(16)));
typedef __bf16 bf8v  __attribute__((ext_vector_type(8)));
typedef float  v8f   __attribute__((ext_vector_type(8)));

#define DIM 384
#define HEADS 12
#define HEAD_DIM 32
#define WS7 7
#define HW 56
#define BATCH 32
#define TOKENS (BATCH * HW * HW)        // 100352
#define ATT_SCALE 0.17677669529663687f  // 32^-0.5
#define BSTRIDE 40                      // padded LDS row stride (bf16 elems)

// ---------------------------------------------------------------------------
// Kernel 1: depthwise 7x7 conv + bias, NCHW in -> NHWC f32 out
// ---------------------------------------------------------------------------
__global__ __launch_bounds__(256) void dwconv_kernel(
    const float* __restrict__ x, const float* __restrict__ w,
    const float* __restrict__ bias, float* __restrict__ x1) {
  size_t idx = (size_t)blockIdx.x * 256 + threadIdx.x;
  if (idx >= (size_t)BATCH * DIM * HW * HW) return;
  int hw = (int)(idx % (HW * HW));
  int c  = (int)((idx / (HW * HW)) % DIM);
  int b  = (int)(idx / ((size_t)DIM * HW * HW));
  int h = hw / HW, wq = hw % HW;
  const float* xin = x + ((size_t)b * DIM + c) * (HW * HW);
  const float* wc = w + c * 49;
  float acc = bias[c];
#pragma unroll
  for (int dh = 0; dh < 7; ++dh) {
    int ih = h + dh - 3;
    if (ih < 0 || ih >= HW) continue;
#pragma unroll
    for (int dw = 0; dw < 7; ++dw) {
      int iw = wq + dw - 3;
      if (iw < 0 || iw >= HW) continue;
      acc += xin[ih * HW + iw] * wc[dh * 7 + dw];
    }
  }
  x1[(((size_t)b * HW + h) * HW + wq) * DIM + c] = acc;
}

// ---------------------------------------------------------------------------
// Kernel 2: LayerNorm over C=384, one wave32 per token; f32 in -> bf16 out
// ---------------------------------------------------------------------------
__global__ __launch_bounds__(256) void ln_kernel(
    const float* __restrict__ x1, const float* __restrict__ g,
    const float* __restrict__ b, __bf16* __restrict__ xn) {
  int lane = threadIdx.x & 31;
  int wave = threadIdx.x >> 5;
  int t = blockIdx.x * 8 + wave;
  if (t >= TOKENS) return;
  const float* row = x1 + (size_t)t * DIM;
  float v[12];
  float sum = 0.f;
#pragma unroll
  for (int i = 0; i < 12; ++i) { v[i] = row[lane + 32 * i]; sum += v[i]; }
#pragma unroll
  for (int o = 16; o > 0; o >>= 1) sum += __shfl_xor(sum, o, 32);
  float mu = sum * (1.f / DIM);
  float var = 0.f;
#pragma unroll
  for (int i = 0; i < 12; ++i) { float d = v[i] - mu; var += d * d; }
#pragma unroll
  for (int o = 16; o > 0; o >>= 1) var += __shfl_xor(var, o, 32);
  float inv = rsqrtf(var * (1.f / DIM) + 1e-6f);
  __bf16* out = xn + (size_t)t * DIM;
#pragma unroll
  for (int i = 0; i < 12; ++i) {
    int c = lane + 32 * i;
    out[c] = (__bf16)((v[i] - mu) * inv * g[c] + b[c]);
  }
}

// ---------------------------------------------------------------------------
// Kernel 3: f32 -> bf16 conversion (weights)
// ---------------------------------------------------------------------------
__global__ __launch_bounds__(256) void cvt_bf16_kernel(
    const float* __restrict__ s, __bf16* __restrict__ d, int n) {
  int i = blockIdx.x * 256 + threadIdx.x;
  if (i < n) d[i] = (__bf16)s[i];
}

// ---------------------------------------------------------------------------
// Kernel 4: WMMA bf16 GEMM  D[M,N] = A[M,K] * W[N,K]^T  (+ epilogue)
// Block = 256 threads = 8 waves; block tile 128(M) x 128(N).
// Waves: 4 in M x 2 in N; each wave computes 32x64 = 8 WMMA accumulators.
// B tile (32K x 128N) staged in LDS, double buffered, register-prefetched:
// global loads for step s+1 are in flight while WMMAs for step s execute.
// LDS row stride padded to 40 bf16 (bank stride 20 mod 64 -> conflict-free).
// A fragment layout (16-bit A, ISA 7.12.2): lane<16 -> row=lane, K {0..7,
// 16..23}; lane>=16 -> row=lane-16, K {8..15, 24..31} within the 32-slab.
// Epilogues: 0=qkv(raw bf16), 1=proj(+bias+res->bf16), 2=pw1(+bias,GELU->bf16),
//            3=pw2(+bias,BN,+x -> f32 NCHW)
// ---------------------------------------------------------------------------
template <int EPI>
__global__ __launch_bounds__(256) void gemm_wmma_kernel(
    const __bf16* __restrict__ A, const __bf16* __restrict__ W,
    const float* __restrict__ bias, const float* __restrict__ resid,
    const float* __restrict__ bnw, const float* __restrict__ bnb,
    const float* __restrict__ bnm, const float* __restrict__ bnv,
    const float* __restrict__ xin, __bf16* __restrict__ out_bf,
    float* __restrict__ out_f, int K, int ldo) {
  __shared__ __align__(16) __bf16 Bs[2][128 * BSTRIDE];

  const int tid  = threadIdx.x;
  const int lane = tid & 31;
  const int wid  = tid >> 5;
  const int wm   = wid & 3;                    // M quadrant (32 rows each)
  const int wn   = wid >> 2;                   // N half (64 cols each)
  const int m0   = blockIdx.x * 128 + wm * 32;
  const int n0b  = blockIdx.y * 128;           // block N base
  const int n0   = n0b + wn * 64;              // wave N base

  const int rlo = lane & 15;
  const int kh  = lane >> 4;

  // cooperative B-tile loader: 2 chunks of 8 bf16 per thread (512 chunks)
  const int nl0 = (tid * 2) >> 2;
  const int kl0 = ((tid * 2) & 3) * 8;
  const int nl1 = (tid * 2 + 1) >> 2;
  const int kl1 = ((tid * 2 + 1) & 3) * 8;

  v8f c[2][4] = {};

  const size_t arow0 = (size_t)(m0 + rlo) * K;        // rows 0..15 of wave
  const size_t arow1 = (size_t)(m0 + 16 + rlo) * K;   // rows 16..31 of wave

  // ---- preload first B tile into LDS stage 0
  {
    bf8v t0 = *reinterpret_cast<const bf8v*>(W + (size_t)(n0b + nl0) * K + kl0);
    bf8v t1 = *reinterpret_cast<const bf8v*>(W + (size_t)(n0b + nl1) * K + kl1);
    *reinterpret_cast<bf8v*>(&Bs[0][nl0 * BSTRIDE + kl0]) = t0;
    *reinterpret_cast<bf8v*>(&Bs[0][nl1 * BSTRIDE + kl1]) = t1;
  }
  __syncthreads();

  const int nsteps = K / 32;
  for (int s = 0; s < nsteps; ++s) {
    const int kb = s * 32;
    const int st = s & 1;

    // register-stage the next B slab (uniform branch; EXEC stays full)
    bf8v nb0 = {}, nb1 = {};
    if (s + 1 < nsteps) {
      nb0 = *reinterpret_cast<const bf8v*>(
          W + (size_t)(n0b + nl0) * K + kb + 32 + kl0);
      nb1 = *reinterpret_cast<const bf8v*>(
          W + (size_t)(n0b + nl1) * K + kb + 32 + kl1);
      __builtin_prefetch(A + arow0 + kb + 32, 0, 1);
      __builtin_prefetch(A + arow1 + kb + 32, 0, 1);
    }

    // ---- A fragments (two 16x32 tiles per wave)
    v16bf a0, a1;
    {
      bf8v l0 = *reinterpret_cast<const bf8v*>(A + arow0 + kb + kh * 8);
      bf8v h0 = *reinterpret_cast<const bf8v*>(A + arow0 + kb + 16 + kh * 8);
      bf8v l1 = *reinterpret_cast<const bf8v*>(A + arow1 + kb + kh * 8);
      bf8v h1 = *reinterpret_cast<const bf8v*>(A + arow1 + kb + 16 + kh * 8);
#pragma unroll
      for (int i = 0; i < 8; ++i) {
        a0[i] = l0[i]; a0[8 + i] = h0[i];
        a1[i] = l1[i]; a1[8 + i] = h1[i];
      }
    }

    // ---- 4 B fragments from LDS, each feeding two WMMAs (a0, a1)
#pragma unroll
    for (int j = 0; j < 4; ++j) {
      const __bf16* bp =
          &Bs[st][(wn * 64 + j * 16 + rlo) * BSTRIDE + kh * 8];
      bf8v lo = *reinterpret_cast<const bf8v*>(bp);
      bf8v hi = *reinterpret_cast<const bf8v*>(bp + 16);
      v16bf bfr;
#pragma unroll
      for (int i = 0; i < 8; ++i) { bfr[i] = lo[i]; bfr[8 + i] = hi[i]; }
      c[0][j] = __builtin_amdgcn_wmma_f32_16x16x32_bf16(
          false, a0, false, bfr, (short)0, c[0][j], false, false);
      c[1][j] = __builtin_amdgcn_wmma_f32_16x16x32_bf16(
          false, a1, false, bfr, (short)0, c[1][j], false, false);
    }

    __syncthreads();
    if (s + 1 < nsteps) {
      *reinterpret_cast<bf8v*>(&Bs[st ^ 1][nl0 * BSTRIDE + kl0]) = nb0;
      *reinterpret_cast<bf8v*>(&Bs[st ^ 1][nl1 * BSTRIDE + kl1]) = nb1;
    }
    __syncthreads();
  }

  // ---- epilogue. C/D layout: VGPR r, lanes0-15 -> M=r, lanes16-31 -> M=r+8
  const int mrow  = m0 + (lane >> 4) * 8;
  const int nbase = n0 + (lane & 15);
#pragma unroll
  for (int i = 0; i < 2; ++i) {
#pragma unroll
    for (int j = 0; j < 4; ++j) {
      int n = nbase + j * 16;
#pragma unroll
      for (int r = 0; r < 8; ++r) {
        int m = mrow + i * 16 + r;
        float v = c[i][j][r];
        if constexpr (EPI == 0) {
          out_bf[(size_t)m * ldo + n] = (__bf16)v;
        } else if constexpr (EPI == 1) {
          v += bias[n] + resid[(size_t)m * DIM + n];
          out_bf[(size_t)m * ldo + n] = (__bf16)v;
        } else if constexpr (EPI == 2) {
          v += bias[n];
          v = 0.5f * v * (1.0f + erff(v * 0.70710678118654752f));
          out_bf[(size_t)m * ldo + n] = (__bf16)v;
        } else {  // EPI == 3: bias + BN(inference) + input residual, NCHW f32
          v += bias[n];
          v = (v - bnm[n]) * rsqrtf(bnv[n] + 1e-5f) * bnw[n] + bnb[n];
          int bb = m / (HW * HW);
          int hw = m % (HW * HW);
          size_t oi = ((size_t)bb * DIM + n) * (HW * HW) + hw;
          out_f[oi] = v + xin[oi];
        }
      }
    }
  }
}

// ---------------------------------------------------------------------------
// Kernel 5: window attention, flash-style online softmax.
// One 64-thread block per (batch, window, head); K/V staged in LDS.
// qkv layout per token: [q(384) | k(384) | v(384)], head h at offset h*32.
// ---------------------------------------------------------------------------
__global__ __launch_bounds__(64) void attn_kernel(
    const __bf16* __restrict__ qkv, __bf16* __restrict__ attn_out) {
  __shared__ __bf16 Ks[49 * HEAD_DIM];
  __shared__ __bf16 Vs[49 * HEAD_DIM];

  int bg = blockIdx.x;
  int b = bg >> 6;            // 64 windows per image (8x8)
  int g = bg & 63;
  int gh = g >> 3, gw = g & 7;
  int head = blockIdx.y;
  int tid = threadIdx.x;

  for (int i = tid; i < 49 * HEAD_DIM; i += 64) {
    int p = i / HEAD_DIM, d = i % HEAD_DIM;
    int himg = gh * WS7 + p / WS7;
    int wimg = gw * WS7 + p % WS7;
    size_t t = ((size_t)b * HW + himg) * HW + wimg;
    const __bf16* base = qkv + t * (3 * DIM) + head * HEAD_DIM + d;
    Ks[i] = base[DIM];
    Vs[i] = base[2 * DIM];
  }
  __syncthreads();

  if (tid >= 49) return;
  int himg = gh * WS7 + tid / WS7;
  int wimg = gw * WS7 + tid % WS7;
  size_t tq = ((size_t)b * HW + himg) * HW + wimg;

  float qv[HEAD_DIM];
  const __bf16* qp = qkv + tq * (3 * DIM) + head * HEAD_DIM;
#pragma unroll
  for (int d = 0; d < HEAD_DIM; ++d) qv[d] = (float)qp[d];

  float m = -1e30f, s = 0.f;
  float acc[HEAD_DIM];
#pragma unroll
  for (int d = 0; d < HEAD_DIM; ++d) acc[d] = 0.f;

  for (int k = 0; k < 49; ++k) {
    float sc = 0.f;
    const __bf16* kr = &Ks[k * HEAD_DIM];
#pragma unroll
    for (int d = 0; d < HEAD_DIM; ++d) sc += qv[d] * (float)kr[d];
    sc *= ATT_SCALE;
    float mn = fmaxf(m, sc);
    float corr = __expf(m - mn);
    float p = __expf(sc - mn);
    s = s * corr + p;
    const __bf16* vr = &Vs[k * HEAD_DIM];
#pragma unroll
    for (int d = 0; d < HEAD_DIM; ++d)
      acc[d] = acc[d] * corr + p * (float)vr[d];
    m = mn;
  }
  float inv = 1.f / s;
  __bf16* op = attn_out + tq * DIM + head * HEAD_DIM;
#pragma unroll
  for (int d = 0; d < HEAD_DIM; ++d) op[d] = (__bf16)(acc[d] * inv);
}

// ---------------------------------------------------------------------------
// Host launcher
// ---------------------------------------------------------------------------
extern "C" void kernel_launch(void* const* d_in, const int* in_sizes, int n_in,
                              void* d_out, int out_size, void* d_ws,
                              size_t ws_size, hipStream_t stream) {
  const float* x      = (const float*)d_in[0];
  const float* dw_w   = (const float*)d_in[1];
  const float* dw_b   = (const float*)d_in[2];
  const float* ln_w   = (const float*)d_in[3];
  const float* ln_b   = (const float*)d_in[4];
  const float* qkv_w  = (const float*)d_in[5];
  const float* proj_w = (const float*)d_in[6];
  const float* proj_b = (const float*)d_in[7];
  const float* pw1_w  = (const float*)d_in[8];
  const float* pw1_b  = (const float*)d_in[9];
  const float* pw2_w  = (const float*)d_in[10];
  const float* pw2_b  = (const float*)d_in[11];
  const float* bn_w   = (const float*)d_in[12];
  const float* bn_b   = (const float*)d_in[13];
  const float* bn_m   = (const float*)d_in[14];
  const float* bn_v   = (const float*)d_in[15];
  float* out = (float*)d_out;

  char* ws = (char*)d_ws;
  const size_t S_X1  = (size_t)TOKENS * DIM * 4;        // 154.1 MB f32 NHWC
  const size_t S_XN  = (size_t)TOKENS * DIM * 2;        //  77.1 MB bf16
  const size_t S_QKV = (size_t)TOKENS * 3 * DIM * 2;    // 231.2 MB bf16
  float*  x1   = (float*)(ws);
  __bf16* xn   = (__bf16*)(ws + S_X1);                  // reused as x2
  __bf16* qkvb = (__bf16*)(ws + S_X1 + S_XN);           // reused as h
  __bf16* attn = (__bf16*)(ws + S_X1 + S_XN + S_QKV);
  char* wbase  = ws + S_X1 + S_XN + S_QKV + S_XN;
  __bf16* wq = (__bf16*)(wbase);                                  // 1152x384
  __bf16* wp = (__bf16*)(wbase + (size_t)1152 * 384 * 2);         //  384x384
  __bf16* w1 = (__bf16*)(wbase + (size_t)(1152 + 384) * 384 * 2); //  768x384
  __bf16* w2 = (__bf16*)(wbase + (size_t)(1152 + 384 + 768) * 384 * 2);
  __bf16* x2 = xn;
  __bf16* h  = qkvb;

  // 1) depthwise conv + bias -> x1 (NHWC f32)
  {
    size_t n = (size_t)BATCH * DIM * HW * HW;
    dwconv_kernel<<<dim3((unsigned)((n + 255) / 256)), dim3(256), 0, stream>>>(
        x, dw_w, dw_b, x1);
  }
  // 2) LayerNorm -> xn (bf16)
  ln_kernel<<<dim3(TOKENS / 8), dim3(256), 0, stream>>>(x1, ln_w, ln_b, xn);
  // 3) weight conversions
  cvt_bf16_kernel<<<dim3((1152 * 384 + 255) / 256), dim3(256), 0, stream>>>(
      qkv_w, wq, 1152 * 384);
  cvt_bf16_kernel<<<dim3((384 * 384 + 255) / 256), dim3(256), 0, stream>>>(
      proj_w, wp, 384 * 384);
  cvt_bf16_kernel<<<dim3((768 * 384 + 255) / 256), dim3(256), 0, stream>>>(
      pw1_w, w1, 768 * 384);
  cvt_bf16_kernel<<<dim3((384 * 768 + 255) / 256), dim3(256), 0, stream>>>(
      pw2_w, w2, 384 * 768);

  const unsigned GM = TOKENS / 128;  // 784
  // 4) qkv = xn @ qkv_w^T   [100352 x 1152]
  gemm_wmma_kernel<0><<<dim3(GM, 1152 / 128), dim3(256), 0, stream>>>(
      xn, wq, nullptr, nullptr, nullptr, nullptr, nullptr, nullptr, nullptr,
      qkvb, nullptr, DIM, 3 * DIM);
  // 5) window attention -> attn (bf16)
  attn_kernel<<<dim3(BATCH * 64, HEADS), dim3(64), 0, stream>>>(qkvb, attn);
  // 6) x2 = attn @ proj_w^T + proj_b + x1   (bf16, overwrites xn)
  gemm_wmma_kernel<1><<<dim3(GM, 384 / 128), dim3(256), 0, stream>>>(
      attn, wp, proj_b, x1, nullptr, nullptr, nullptr, nullptr, nullptr,
      x2, nullptr, DIM, DIM);
  // 7) h = GELU(x2 @ pw1_w^T + pw1_b)   (bf16, overwrites qkv)
  gemm_wmma_kernel<2><<<dim3(GM, 768 / 128), dim3(256), 0, stream>>>(
      x2, w1, pw1_b, nullptr, nullptr, nullptr, nullptr, nullptr, nullptr,
      h, nullptr, DIM, 2 * DIM);
  // 8) out = x + BN(h @ pw2_w^T + pw2_b)   (f32 NCHW)
  gemm_wmma_kernel<3><<<dim3(GM, 384 / 128), dim3(256), 0, stream>>>(
      h, w2, pw2_b, nullptr, bn_w, bn_b, bn_m, bn_v, x,
      nullptr, out, 2 * DIM, DIM);
}
